// SVAE_54674933678838
// MI455X (gfx1250) — compile-verified, matrix operand order checked by reading, not run on hardware
//
#include <hip/hip_runtime.h>
#include <hip/hip_bf16.h>

// Problem constants (from reference)
#define V_  32000
#define H_  512
#define L_  256
#define B_  16
#define TE_ 512
#define TD_ 256

typedef __attribute__((ext_vector_type(16))) __bf16 v16bf;
typedef __attribute__((ext_vector_type(8)))  __bf16 v8bf;
typedef __attribute__((ext_vector_type(8)))  float  v8f;

// Probe for the gfx1250 async global->LDS copy path (ASYNCcnt).
#if defined(__gfx1250__) && __has_builtin(__builtin_amdgcn_global_load_async_to_lds_b128)
#define USE_ASYNC_LDS 1
typedef int v4i_ __attribute__((vector_size(16)));
typedef __attribute__((address_space(1))) v4i_* gptr_v4i;   // global
typedef __attribute__((address_space(3))) v4i_* lptr_v4i;   // LDS
#else
#define USE_ASYNC_LDS 0
#endif

__device__ __forceinline__ unsigned short f2bf(float f) {
  union { float f; unsigned u; } x; x.f = f;
  unsigned r = x.u + 0x7FFFu + ((x.u >> 16) & 1u);   // round-to-nearest-even
  return (unsigned short)(r >> 16);
}

// Load one lane's A/B fragment for v_wmma_f32_16x16x32_bf16.
// Source is row-major with the lane's row at p (K-contiguous).
// Lanes 0-15 hold K = {0..7, 16..23}, lanes 16-31 K = {8..15, 24..31};
// caller passes p already offset by half*8 so we read +0 and +16.
__device__ __forceinline__ v16bf load_frag16(const unsigned short* p) {
  v8bf lo = *reinterpret_cast<const v8bf*>(p);
  v8bf hi = *reinterpret_cast<const v8bf*>(p + 16);
  v16bf f;
#pragma unroll
  for (int i = 0; i < 8; ++i) { f[i] = lo[i]; f[i + 8] = hi[i]; }
  return f;
}

__device__ __forceinline__ v8f wmma_bf16(v16bf a, v16bf b, v8f c) {
  return __builtin_amdgcn_wmma_f32_16x16x32_bf16(
      /*neg_a=*/false, a, /*neg_b=*/false, b,
      /*c_mod=*/(short)0, c, /*reuse_a=*/false, /*reuse_b=*/false);
}

__device__ __forceinline__ float sigmoidf_(float x) {
  return 1.0f / (1.0f + __expf(-x));
}

// ---------------------------------------------------------------------------
// fp32 -> bf16 convert (grid stride)
// ---------------------------------------------------------------------------
__global__ void f32_to_bf16_kernel(const float* __restrict__ src,
                                   unsigned short* __restrict__ dst,
                                   long long n) {
  long long i  = (long long)blockIdx.x * blockDim.x + threadIdx.x;
  long long st = (long long)gridDim.x * blockDim.x;
  for (; i < n; i += st) dst[i] = f2bf(src[i]);
}

// ---------------------------------------------------------------------------
// Embedding gather -> bf16 rows
// ---------------------------------------------------------------------------
__global__ void gather_emb_bf16_kernel(const float* __restrict__ emb,
                                       const int* __restrict__ ids,
                                       unsigned short* __restrict__ dst,
                                       long long n /* tokens*H */) {
  long long i  = (long long)blockIdx.x * blockDim.x + threadIdx.x;
  long long st = (long long)gridDim.x * blockDim.x;
  for (; i < n; i += st) {
    long long tok = i / H_;
    int c = (int)(i - tok * H_);
    dst[i] = f2bf(emb[(size_t)ids[tok] * H_ + c]);
  }
}

// ---------------------------------------------------------------------------
// C[M,N] (f32) = A[M,K] (bf16, row-major) * W[N,K]^T (bf16, row-major) + bias
// 2D grid: blockIdx.y = M/16 tile, blockIdx.x covers 8 n-groups of 64 cols.
// The block's 16 x K A-strip is staged once into LDS (async copy if the
// gfx1250 builtin exists) and shared by all 8 waves via ds_load_b128; each
// wave streams its own 64 W rows from global (L2-resident) into 4 WMMA accums.
// Requires M%16==0, N%64==0, K%32==0 (true at every call site).
// ---------------------------------------------------------------------------
__global__ void gemm_bf16_wmma_kernel(const unsigned short* __restrict__ A,
                                      const unsigned short* __restrict__ W,
                                      const float* __restrict__ bias,
                                      float* __restrict__ C,
                                      int M, int N, int K) {
  extern __shared__ unsigned short aTile[];   // 16 * K bf16
  const int tid  = threadIdx.x;
  const int lane = tid & 31;
  const int wib  = tid >> 5;                  // wave in block (0..7)
  const int m0   = blockIdx.y << 4;
  const int ng   = ((blockIdx.x * (blockDim.x >> 5)) + wib) << 6;

  // ---- stage A strip (rows m0..m0+15, contiguous 16*K elements) into LDS ----
  {
    const unsigned short* abase = A + (size_t)m0 * K;
    const int chunks = (16 * K) >> 3;         // 16-byte chunks
    for (int c = tid; c < chunks; c += blockDim.x) {
#if USE_ASYNC_LDS
      __builtin_amdgcn_global_load_async_to_lds_b128(
          (gptr_v4i)(const_cast<unsigned short*>(abase) + (size_t)c * 8),
          (lptr_v4i)(aTile + c * 8), 0, 0);
#else
      reinterpret_cast<v8bf*>(aTile)[c] =
          reinterpret_cast<const v8bf*>(abase)[c];
#endif
    }
#if USE_ASYNC_LDS
#if __has_builtin(__builtin_amdgcn_s_wait_asynccnt)
    __builtin_amdgcn_s_wait_asynccnt(0);
#else
    asm volatile("s_wait_asynccnt 0" ::: "memory");
#endif
#endif
    __syncthreads();
  }

  if (ng >= N) return;                        // N%64==0, so ng<N => full group

  const int half = lane >> 4;
  const int r16  = lane & 15;

  const unsigned short* arow = aTile + r16 * K + half * 8;   // LDS
  const unsigned short* w0   = W + (size_t)(ng      + r16) * K + half * 8;
  const unsigned short* w1   = W + (size_t)(ng + 16 + r16) * K + half * 8;
  const unsigned short* w2   = W + (size_t)(ng + 32 + r16) * K + half * 8;
  const unsigned short* w3   = W + (size_t)(ng + 48 + r16) * K + half * 8;

  v8f acc0 = {}, acc1 = {}, acc2 = {}, acc3 = {};
  for (int k0 = 0; k0 < K; k0 += 32) {
    __builtin_prefetch(w0 + k0 + 128, 0, 0);   // stream next weight chunk
    v16bf a  = load_frag16(arow + k0);         // ds_load path (shared strip)
    v16bf b0 = load_frag16(w0 + k0);
    v16bf b1 = load_frag16(w1 + k0);
    v16bf b2 = load_frag16(w2 + k0);
    v16bf b3 = load_frag16(w3 + k0);
    acc0 = wmma_bf16(a, b0, acc0);
    acc1 = wmma_bf16(a, b1, acc1);
    acc2 = wmma_bf16(a, b2, acc2);
    acc3 = wmma_bf16(a, b3, acc3);
  }

  // D layout: VGPR r -> row m0 + r + half*8, column = tile_n0 + (lane&15)
#pragma unroll
  for (int j = 0; j < 4; ++j) {
    const v8f acc = (j == 0) ? acc0 : (j == 1) ? acc1 : (j == 2) ? acc2 : acc3;
    const int ncol = ng + j * 16 + r16;
    const float bv = bias ? bias[ncol] : 0.0f;
#pragma unroll
    for (int r = 0; r < 8; ++r) {
      C[(size_t)(m0 + r + half * 8) * N + ncol] = acc[r] + bv;
    }
  }
}

// ---------------------------------------------------------------------------
// GRU scan (one persistent workgroup, 512 threads = 16 waves).
// gx (B*T x 3Hout, already + bih) is precomputed.  Per step:
//   gh = h_bf16(16 x Hout) @ Whh^T  via WMMA (M=16 == batch),
//   then VALU gate math updates h (kept in registers, bf16 mirror in LDS).
// Dynamic LDS: float gh[16*3Hout] + ushort hbf[16*Hout].
// ---------------------------------------------------------------------------
__global__ void gru_scan_kernel(const float* __restrict__ gx,
                                const unsigned short* __restrict__ Whh, // 3Hout x Hout bf16
                                const float* __restrict__ bhh,          // 3Hout
                                const int* __restrict__ len,            // B
                                const float* __restrict__ h0,           // B x Hout or null
                                float* __restrict__ out_f32,            // B x T x Hout or null
                                unsigned short* __restrict__ out_bf16,  // B x T x Hout or null
                                float* __restrict__ h_last,             // B x Hout or null
                                int T, int Hout) {
  extern __shared__ char smem[];
  const int N3 = 3 * Hout;
  float* ghs = reinterpret_cast<float*>(smem);                          // 16 * N3
  unsigned short* hbf =
      reinterpret_cast<unsigned short*>(smem + (size_t)16 * N3 * sizeof(float)); // 16*Hout

  const int tid    = threadIdx.x;
  const int nth    = blockDim.x;            // 512
  const int lane   = tid & 31;
  const int wave   = tid >> 5;
  const int nwaves = nth >> 5;              // 16
  const int half   = lane >> 4;
  const int r16    = lane & 15;
  const int ntiles = N3 >> 4;               // 96 (enc) / 48 (dec)
  const int tpw    = ntiles / nwaves;       // 6 / 3
  const int E      = (B_ * Hout) / nth;     // 16 / 8 elements per thread

  float hreg[16];
  for (int j = 0; j < E; ++j) {
    int e = tid * E + j;
    float h = h0 ? h0[e] : 0.0f;
    hreg[j] = h;
    hbf[e]  = f2bf(h);
  }
  __syncthreads();

  for (int t = 0; t < T; ++t) {
    // ---- gh = h @ Whh^T (16 x N3) via WMMA, A from LDS, B streamed ----
    for (int tt = 0; tt < tpw; ++tt) {
      const int n0 = (wave * tpw + tt) << 4;
      v8f acc = {};
      const unsigned short* hrow = hbf + r16 * Hout + half * 8;
      const unsigned short* wrow = Whh + (size_t)(n0 + r16) * Hout + half * 8;
      for (int k0 = 0; k0 < Hout; k0 += 32) {
        v16bf a = load_frag16(hrow + k0);          // ds_load path
        v16bf b = load_frag16(wrow + k0);          // global path
        acc = wmma_bf16(a, b, acc);
      }
#pragma unroll
      for (int r = 0; r < 8; ++r)
        ghs[(size_t)(r + half * 8) * N3 + n0 + r16] = acc[r];
    }
    __syncthreads();

    // ---- gate combine + state update ----
    for (int j = 0; j < E; ++j) {
      const int e = tid * E + j;
      const int b = e / Hout;
      const int u = e - b * Hout;
      const float* gxr = gx + (size_t)(b * T + t) * N3;
      const float ghr = ghs[(size_t)b * N3 + u]            + bhh[u];
      const float ghz = ghs[(size_t)b * N3 + Hout + u]     + bhh[Hout + u];
      const float ghn = ghs[(size_t)b * N3 + 2 * Hout + u] + bhh[2 * Hout + u];
      const float r = sigmoidf_(gxr[u] + ghr);
      const float z = sigmoidf_(gxr[Hout + u] + ghz);
      const float n = tanhf(gxr[2 * Hout + u] + r * ghn);
      const float hn = (1.0f - z) * n + z * hreg[j];
      const bool  v  = (t < len[b]);
      const float hk = v ? hn : hreg[j];
      const float ov = v ? hn : 0.0f;
      hreg[j] = hk;
      hbf[e]  = f2bf(hk);
      if (out_f32)  out_f32 [((size_t)b * T + t) * Hout + u] = ov;
      if (out_bf16) out_bf16[((size_t)b * T + t) * Hout + u] = f2bf(ov);
    }
    __syncthreads();
  }

  if (h_last)
    for (int j = 0; j < E; ++j) h_last[tid * E + j] = hreg[j];
}

// ---------------------------------------------------------------------------
// Attention: scores vs last/first state, masked softmax, semantic pooling,
// h_enc = concat(last, semantic).  One block per batch row.
// ---------------------------------------------------------------------------
__device__ float blk_reduce_max(float v, float* red) {
  int tid = threadIdx.x;
  red[tid] = v; __syncthreads();
  for (int s = 128; s > 0; s >>= 1) {
    if (tid < s) red[tid] = fmaxf(red[tid], red[tid + s]);
    __syncthreads();
  }
  float r = red[0]; __syncthreads();
  return r;
}
__device__ float blk_reduce_sum(float v, float* red) {
  int tid = threadIdx.x;
  red[tid] = v; __syncthreads();
  for (int s = 128; s > 0; s >>= 1) {
    if (tid < s) red[tid] += red[tid + s];
    __syncthreads();
  }
  float r = red[0]; __syncthreads();
  return r;
}

__global__ void attention_kernel(const float* __restrict__ enc_out, // B x TE x H
                                 const float* __restrict__ h_last,  // B x H
                                 const int* __restrict__ enc_len,   // B
                                 float* __restrict__ h_enc) {       // B x 2H
  __shared__ float sfw[TE_];
  __shared__ float sbw[TE_];
  __shared__ float red[256];
  const int b   = blockIdx.x;
  const int tid = threadIdx.x;
  const float* ob    = enc_out + (size_t)b * TE_ * H_;
  const float* last  = h_last + (size_t)b * H_;
  const float* first = ob;                         // out[b, 0, :]
  const int len = enc_len[b];

  for (int t = tid; t < TE_; t += blockDim.x) {
    float afw = 0.0f, abw = 0.0f;
    const float* ot = ob + (size_t)t * H_;
    for (int d = 0; d < H_; ++d) {
      const float o = ot[d];
      afw += o * last[d];
      abw += o * first[d];
    }
    sfw[t] = afw; sbw[t] = abw;
  }
  __syncthreads();

  float mfw = -1e30f, mbw = -1e30f;
  for (int t = tid; t < TE_; t += blockDim.x)
    if (t < len) { mfw = fmaxf(mfw, sfw[t]); mbw = fmaxf(mbw, sbw[t]); }
  mfw = blk_reduce_max(mfw, red);
  mbw = blk_reduce_max(mbw, red);

  float s1 = 0.0f, s2 = 0.0f;
  for (int t = tid; t < TE_; t += blockDim.x) {
    if (t < len) {
      float e1 = __expf(sfw[t] - mfw); sfw[t] = e1; s1 += e1;
      float e2 = __expf(sbw[t] - mbw); sbw[t] = e2; s2 += e2;
    } else { sfw[t] = 0.0f; sbw[t] = 0.0f; }
  }
  s1 = blk_reduce_sum(s1, red);
  s2 = blk_reduce_sum(s2, red);

  for (int t = tid; t < TE_; t += blockDim.x)
    sfw[t] = (t < len) ? 0.5f * (sfw[t] / s1 + sbw[t] / s2) : 0.0f;
  __syncthreads();

  for (int d = tid; d < H_; d += blockDim.x) {
    float s = 0.0f;
    for (int t = 0; t < len; ++t) s += sfw[t] * ob[(size_t)t * H_ + d];
    h_enc[(size_t)b * 2 * H_ + d]      = last[d];
    h_enc[(size_t)b * 2 * H_ + H_ + d] = s;
  }
}

// ---------------------------------------------------------------------------
// mu / sigma / z (reparameterization).  One thread per (b, l).
// ---------------------------------------------------------------------------
__global__ void latent_kernel(const float* __restrict__ h_enc,   // B x 2H
                              const float* __restrict__ W_mu,    // L x 2H
                              const float* __restrict__ b_mu,
                              const float* __restrict__ W_ls,    // L x 2H
                              const float* __restrict__ b_ls,
                              const float* __restrict__ noise,   // B x L
                              float* __restrict__ mu_out,
                              float* __restrict__ sigma_out,
                              float* __restrict__ z_out) {
  const int i = blockIdx.x * blockDim.x + threadIdx.x;
  if (i >= B_ * L_) return;
  const int b = i / L_;
  const int l = i - b * L_;
  const float* he = h_enc + (size_t)b * 2 * H_;
  const float* wm = W_mu + (size_t)l * 2 * H_;
  const float* wl = W_ls + (size_t)l * 2 * H_;
  float sm = b_mu[l], sl = b_ls[l];
  for (int d = 0; d < 2 * H_; ++d) {
    const float h = he[d];
    sm += h * wm[d];
    sl += h * wl[d];
  }
  const float sig = __expf(sl);
  mu_out[i]    = sm;
  sigma_out[i] = sig;
  z_out[i]     = sm + sig * noise[i];
}

__global__ void maxlen_kernel(const int* __restrict__ dec_len, float* __restrict__ out) {
  if (blockIdx.x == 0 && threadIdx.x == 0) {
    int m = dec_len[0];
    for (int i = 1; i < B_; ++i) m = max(m, dec_len[i]);
    *out = (float)m;
  }
}

// ---------------------------------------------------------------------------
extern "C" void kernel_launch(void* const* d_in, const int* in_sizes, int n_in,
                              void* d_out, int out_size, void* d_ws, size_t ws_size,
                              hipStream_t stream) {
  (void)in_sizes; (void)n_in; (void)out_size; (void)ws_size;

  const float* emb      = (const float*)d_in[0];
  const float* enc_Wih  = (const float*)d_in[1];
  const float* enc_Whh  = (const float*)d_in[2];
  const float* enc_bih  = (const float*)d_in[3];
  const float* enc_bhh  = (const float*)d_in[4];
  const float* dec_Wih  = (const float*)d_in[5];
  const float* dec_Whh  = (const float*)d_in[6];
  const float* dec_bih  = (const float*)d_in[7];
  const float* dec_bhh  = (const float*)d_in[8];
  const float* W_mu     = (const float*)d_in[9];
  const float* b_mu     = (const float*)d_in[10];
  const float* W_ls     = (const float*)d_in[11];
  const float* b_ls     = (const float*)d_in[12];
  const float* W_fc     = (const float*)d_in[13];
  const float* b_fc     = (const float*)d_in[14];
  const float* noise    = (const float*)d_in[15];
  const int*   enc_ids  = (const int*)d_in[16];
  const int*   enc_len  = (const int*)d_in[17];
  const int*   dec_ids  = (const int*)d_in[18];
  const int*   dec_len  = (const int*)d_in[19];
  float* out = (float*)d_out;

  // -------- workspace carve-out (256B aligned) --------
  char* ws = (char*)d_ws;
  size_t off = 0;
  auto alloc = [&](size_t bytes) -> char* {
    char* p = ws + off;
    off = (off + bytes + 255) & ~(size_t)255;
    return p;
  };
  unsigned short* enc_Wih_bf = (unsigned short*)alloc((size_t)3 * H_ * H_ * 2);
  unsigned short* enc_Whh_bf = (unsigned short*)alloc((size_t)3 * H_ * H_ * 2);
  unsigned short* dec_Wih_bf = (unsigned short*)alloc((size_t)3 * L_ * H_ * 2);
  unsigned short* dec_Whh_bf = (unsigned short*)alloc((size_t)3 * L_ * L_ * 2);
  unsigned short* W_fc_bf    = (unsigned short*)alloc((size_t)V_ * L_ * 2);
  unsigned short* x_enc_bf   = (unsigned short*)alloc((size_t)B_ * TE_ * H_ * 2);
  unsigned short* x_dec_bf   = (unsigned short*)alloc((size_t)B_ * TD_ * H_ * 2);
  float*          gx_enc     = (float*)alloc((size_t)B_ * TE_ * 3 * H_ * 4);
  float*          gx_dec     = (float*)alloc((size_t)B_ * TD_ * 3 * L_ * 4);
  float*          enc_out_b  = (float*)alloc((size_t)B_ * TE_ * H_ * 4);
  float*          h_last     = (float*)alloc((size_t)B_ * H_ * 4);
  float*          h_enc      = (float*)alloc((size_t)B_ * 2 * H_ * 4);
  float*          z_lat      = (float*)alloc((size_t)B_ * L_ * 4);
  unsigned short* dec_out_bf = (unsigned short*)alloc((size_t)B_ * TD_ * L_ * 2);

  // -------- output offsets: (logits, dec_max_len, mu, sigma) --------
  const long long OFF_MAXLEN = (long long)B_ * TD_ * V_;
  const long long OFF_MU     = OFF_MAXLEN + 1;
  const long long OFF_SIGMA  = OFF_MU + (long long)B_ * L_;

  auto cvt = [&](const float* s, unsigned short* d, long long n) {
    int blocks = (int)((n + 255) / 256);
    if (blocks > 2048) blocks = 2048;
    f32_to_bf16_kernel<<<blocks, 256, 0, stream>>>(s, d, n);
  };

  // 1) weight conversions
  cvt(enc_Wih, enc_Wih_bf, (long long)3 * H_ * H_);
  cvt(enc_Whh, enc_Whh_bf, (long long)3 * H_ * H_);
  cvt(dec_Wih, dec_Wih_bf, (long long)3 * L_ * H_);
  cvt(dec_Whh, dec_Whh_bf, (long long)3 * L_ * L_);
  cvt(W_fc,    W_fc_bf,    (long long)V_ * L_);

  // 2) embedding gathers (row = b*T + t, K-contiguous)
  {
    long long n = (long long)B_ * TE_ * H_;
    gather_emb_bf16_kernel<<<2048, 256, 0, stream>>>(emb, enc_ids, x_enc_bf, n);
    n = (long long)B_ * TD_ * H_;
    gather_emb_bf16_kernel<<<2048, 256, 0, stream>>>(emb, dec_ids, x_dec_bf, n);
  }

  auto gemm = [&](const unsigned short* A, const unsigned short* Wm,
                  const float* bias, float* Cm, int M, int N, int K) {
    const int n4 = N / 64;
    dim3 grid((n4 + 7) / 8, M / 16);
    size_t shmem = (size_t)16 * K * sizeof(unsigned short);
    gemm_bf16_wmma_kernel<<<grid, 256, shmem, stream>>>(A, Wm, bias, Cm, M, N, K);
  };

  // 3) gx GEMMs (time-parallel input projections): C = A @ W^T + bias
  gemm(x_enc_bf, enc_Wih_bf, enc_bih, gx_enc, B_ * TE_, 3 * H_, H_); // 8192x1536x512
  gemm(x_dec_bf, dec_Wih_bf, dec_bih, gx_dec, B_ * TD_, 3 * L_, H_); // 4096x768x512

  // 4) encoder scan (1 persistent WG, 16 waves, WMMA M=16=batch)
  {
    size_t shmem = (size_t)16 * 3 * H_ * sizeof(float) + (size_t)16 * H_ * 2;
    gru_scan_kernel<<<1, 512, shmem, stream>>>(gx_enc, enc_Whh_bf, enc_bhh, enc_len,
                                               nullptr, enc_out_b, nullptr, h_last,
                                               TE_, H_);
  }

  // 5) attention + h_enc
  attention_kernel<<<B_, 256, 0, stream>>>(enc_out_b, h_last, enc_len, h_enc);

  // 6) mu / sigma / z  (mu, sigma straight into d_out)
  latent_kernel<<<(B_ * L_ + 255) / 256, 256, 0, stream>>>(
      h_enc, W_mu, b_mu, W_ls, b_ls, noise,
      out + OFF_MU, out + OFF_SIGMA, z_lat);

  // 7) decoder scan (writes masked dec_out as bf16 for the logits GEMM)
  {
    size_t shmem = (size_t)16 * 3 * L_ * sizeof(float) + (size_t)16 * L_ * 2;
    gru_scan_kernel<<<1, 512, shmem, stream>>>(gx_dec, dec_Whh_bf, dec_bhh, dec_len,
                                               z_lat, nullptr, dec_out_bf, nullptr,
                                               TD_, L_);
  }

  // 8) logits GEMM: (B*TD x L) @ (L x V) + b_fc  -> d_out  (dominant, 67.1 GFLOP)
  gemm(dec_out_bf, W_fc_bf, b_fc, out, B_ * TD_, V_, L_);            // 4096x32000x256

  // 9) dec_max_len
  maxlen_kernel<<<1, 32, 0, stream>>>(dec_len, out + OFF_MAXLEN);
}
